// PatchEmbed_44736379355643
// MI455X (gfx1250) — compile-verified, matrix operand order checked by reading, not run on hardware
//
#include <hip/hip_runtime.h>

// ---------------- types for WMMA ----------------
typedef __attribute__((ext_vector_type(16))) __bf16 v16bf;
typedef __attribute__((ext_vector_type(8)))  float  v8f;

union FragU { uint4 q[2]; v16bf v; };

// ---------------- problem constants ----------------
#define IMGW   384
#define PATCH  16
#define CIN    3
#define EMB    768        // also K = CIN*PATCH*PATCH = 768
#define BATCH  64
#define FS     24         // IMG/PATCH
#define PPB    576        // FS*FS patches per image
#define MROWS  36864      // BATCH*PPB
#define OSZ    28311552   // MROWS*EMB  (one image's output floats)
#define EPSV   0.0001f

// LDS tile stride (bf16 elements), padded for bank stagger, 16B-aligned rows
#define LDA 40

__device__ __forceinline__ unsigned short f2bf(float f) {
    unsigned int u = __float_as_uint(f);
    unsigned int r = u + 0x7FFFu + ((u >> 16) & 1u);
    return (unsigned short)(r >> 16);
}

// ---------------- 1. weight pack: fp32 -> bf16 (layout already [e][k]) ----
__global__ void convw_kernel(const float* __restrict__ W, unsigned short* __restrict__ Wt) {
    int i = blockIdx.x * blockDim.x + threadIdx.x;   // quad index
    if (i >= (EMB * EMB) / 4) return;
    int idx = i * 4;
    float4 v = *(const float4*)(W + idx);
    ushort4 o;
    o.x = f2bf(v.x); o.y = f2bf(v.y); o.z = f2bf(v.z); o.w = f2bf(v.w);
    *(ushort4*)(Wt + idx) = o;
}

// ---------------- 2. im2col + fp32 -> bf16 -------------------------------
__global__ void im2col_kernel(const float* __restrict__ x, const float* __restrict__ x2,
                              unsigned short* __restrict__ A1, unsigned short* __restrict__ A2) {
    int i = blockIdx.x * blockDim.x + threadIdx.x;   // quad index: MROWS*192 quads
    if (i >= MROWS * 192) return;
    int row = i / 192;
    int kq  = (i % 192) * 4;                  // k = c*256 + dy*16 + dx, dx%4==0
    int b  = row / PPB,  p  = row % PPB;
    int py = p / FS,     px = p % FS;
    int c  = kq >> 8;
    int r  = kq & 255;
    int dy = r >> 4, dx = r & 15;
    const float*     src = (blockIdx.y == 0) ? x  : x2;
    unsigned short*  dst = (blockIdx.y == 0) ? A1 : A2;
    size_t si = ((((size_t)b * CIN + c) * IMGW) + py * PATCH + dy) * IMGW + px * PATCH + dx;
    float4 v = *(const float4*)(src + si);
    ushort4 o;
    o.x = f2bf(v.x); o.y = f2bf(v.y); o.z = f2bf(v.z); o.w = f2bf(v.w);
    *(ushort4*)(dst + (size_t)row * EMB + kq) = o;
}

// ---------------- 3. boxes / occlusion / skip / bbox-update ---------------
__device__ __forceinline__ float4 upd_bbox(float x0, float y0, float x2, float y2,
                                           float q0, float q1, float q2, float q3) {
    bool condA = (x0 >= q0) && (x2 <= q2);
    bool subA1 = (q1 <= y0) && (q3 > y0);
    bool a1 = condA && subA1;
    bool a2 = condA && !subA1 && (q3 >= y2) && (q1 < y2);
    bool condB = !condA && (y0 >= q1) && (y2 <= q3);
    bool subB1 = (q0 <= x0) && (q2 > x0);
    bool b1 = condB && subB1;
    bool b2 = condB && !subB1 && (q2 >= x2) && (q0 < x2);
    float4 o;
    o.y = a1 ? q3 : y0;
    o.w = a2 ? q1 : y2;
    o.x = b1 ? q2 : x0;
    o.z = b2 ? q0 : x2;
    return o;
}

__global__ void box_kernel(const float* __restrict__ gt1, const float* __restrict__ gt2,
                           float* __restrict__ g1s, float* __restrict__ g2s,
                           int* __restrict__ sk1, int* __restrict__ sk2,
                           float* __restrict__ gout1, float* __restrict__ gout2) {
    int b = threadIdx.x;
    if (b >= BATCH) return;
    float a0 = gt1[b*4+0] * (float)IMGW, a1 = gt1[b*4+1] * (float)IMGW;
    float a2 = gt1[b*4+2] * (float)IMGW, a3 = gt1[b*4+3] * (float)IMGW;
    float c0 = gt2[b*4+0] * (float)IMGW, c1 = gt2[b*4+1] * (float)IMGW;
    float c2 = gt2[b*4+2] * (float)IMGW, c3 = gt2[b*4+3] * (float)IMGW;
    const float inv = 1.0f / (float)PATCH;
    float x0 = rintf(a0 * inv),            y0 = rintf(a1 * inv);
    float x2 = rintf((a0 + a2 - 1.f)*inv), y2 = rintf((a1 + a3 - 1.f)*inv);
    float q0 = rintf(c0 * inv),            q1 = rintf(c1 * inv);
    float q2 = rintf((c0 + c2 - 1.f)*inv), q3 = rintf((c1 + c3 - 1.f)*inv);

    float ixmin = fmaxf(x0, q0), iymin = fmaxf(y0, q1);
    float ixmax = fminf(x2, q2), iymax = fminf(y2, q3);
    float iv   = fmaxf(iymax - iymin, 0.f) * fmaxf(ixmax - ixmin, 0.f);
    float vol1 = (x2 - x0) * (y2 - y0);
    float vol2 = (q2 - q0) * (q3 - q1);
    float occ1 = (iv + 1e-8f) / (vol1 + 1e-8f);
    float occ2 = (iv + 1e-8f) / (vol2 + 1e-8f);
    const float fsf = (float)FS;
    bool s1 = (occ1 > 0.5f) || (q3 <= q1) || (q2 <= q0) || (q0 < 0.f) || (q1 < 0.f) || (q0 >= fsf) || (q1 >= fsf);
    bool s2 = (occ2 > 0.5f) || (y2 <= y0) || (x2 <= x0) || (x0 < 0.f) || (y0 < 0.f) || (x0 >= fsf) || (y0 >= fsf);

    float4 n1 = upd_bbox(x0, y0, x2, y2, q0, q1, q2, q3);
    float4 n2 = upd_bbox(q0, q1, q2, q3, x0, y0, x2, y2);
    if (s1) { n1.x = x0; n1.y = y0; n1.z = x2; n1.w = y2; }
    if (s2) { n2.x = q0; n2.y = q1; n2.z = q2; n2.w = q3; }

    gout1[b*4+0] = n1.x; gout1[b*4+1] = n1.y; gout1[b*4+2] = n1.z - n1.x; gout1[b*4+3] = n1.w - n1.y;
    gout2[b*4+0] = n2.x; gout2[b*4+1] = n2.y; gout2[b*4+2] = n2.z - n2.x; gout2[b*4+3] = n2.w - n2.y;
    g1s[b*4+0] = x0; g1s[b*4+1] = y0; g1s[b*4+2] = x2; g1s[b*4+3] = y2;
    g2s[b*4+0] = q0; g2s[b*4+1] = q1; g2s[b*4+2] = q2; g2s[b*4+3] = q3;
    sk1[b] = s1 ? 1 : 0;
    sk2[b] = s2 ? 1 : 0;
}

// ---------------- 4. bf16 WMMA GEMM: out[row][e] = A[row][:] . Wt[e][:] + b[e]
// Block: 256 threads = 8 wave32, wave grid 4(M) x 2(N); BM=BN=128, BK=32.
// Each wave: 32x64 = 2x4 tiles of 16x16 -> 8 v8f accumulators.
__global__ __launch_bounds__(256) void gemm_kernel(const unsigned short* __restrict__ A1,
                                                   const unsigned short* __restrict__ A2,
                                                   const unsigned short* __restrict__ Wt,
                                                   const float* __restrict__ bias,
                                                   float* __restrict__ out) {
    __shared__ unsigned short sA[128 * LDA];
    __shared__ unsigned short sB[128 * LDA];

    const unsigned short* A = blockIdx.z ? A2 : A1;
    float* o = out + (size_t)blockIdx.z * (size_t)OSZ;

    const int tid   = threadIdx.x;
    const int lane  = tid & 31;
    const int wid   = tid >> 5;
    const int waveM = wid & 3;        // 0..3 -> 32-row slabs
    const int waveN = wid >> 2;       // 0..1 -> 64-col slabs
    const int rowBase = blockIdx.y * 128;
    const int colBase = blockIdx.x * 128;

    v8f acc[2][4];
#pragma unroll
    for (int i = 0; i < 2; ++i)
#pragma unroll
        for (int j = 0; j < 4; ++j)
#pragma unroll
            for (int r = 0; r < 8; ++r) acc[i][j][r] = 0.f;

    // cooperative tile loads: thread -> (row, 16-elem segment)
    const int ldRow = tid >> 1;            // 0..127
    const int ldSeg = (tid & 1) * 16;      // 0 or 16 bf16 elements
    const unsigned short* gA = A  + (size_t)(rowBase + ldRow) * EMB + ldSeg;
    const unsigned short* gB = Wt + (size_t)(colBase + ldRow) * EMB + ldSeg;
    unsigned short* lA = sA + ldRow * LDA + ldSeg;
    unsigned short* lB = sB + ldRow * LDA + ldSeg;

    const int rsel = lane & 15;
    const int off  = (lane >> 4) * 8;      // ISA 16-bit A-fragment K-offset per lane half

    for (int k0 = 0; k0 < EMB; k0 += 32) {
        __syncthreads();
        uint4 va0 = *(const uint4*)(gA + k0);
        uint4 va1 = *(const uint4*)(gA + k0 + 8);
        uint4 vb0 = *(const uint4*)(gB + k0);
        uint4 vb1 = *(const uint4*)(gB + k0 + 8);
        if (k0 + 32 < EMB) {
            __builtin_prefetch(gA + k0 + 32, 0, 1);   // global_prefetch_b8
            __builtin_prefetch(gB + k0 + 32, 0, 1);
        }
        *(uint4*)lA       = va0;
        *(uint4*)(lA + 8) = va1;
        *(uint4*)lB       = vb0;
        *(uint4*)(lB + 8) = vb1;
        __syncthreads();

        v16bf afrag[2], bfrag[4];
#pragma unroll
        for (int i = 0; i < 2; ++i) {
            const unsigned short* pa = sA + (waveM * 32 + i * 16 + rsel) * LDA + off;
            FragU u;
            u.q[0] = *(const uint4*)pa;          // K = off .. off+7
            u.q[1] = *(const uint4*)(pa + 16);   // K = off+16 .. off+23
            afrag[i] = u.v;
        }
#pragma unroll
        for (int j = 0; j < 4; ++j) {
            const unsigned short* pb = sB + (waveN * 64 + j * 16 + rsel) * LDA + off;
            FragU u;
            u.q[0] = *(const uint4*)pb;
            u.q[1] = *(const uint4*)(pb + 16);
            bfrag[j] = u.v;
        }
#pragma unroll
        for (int i = 0; i < 2; ++i)
#pragma unroll
            for (int j = 0; j < 4; ++j)
                acc[i][j] = __builtin_amdgcn_wmma_f32_16x16x32_bf16(
                    false, afrag[i], false, bfrag[j], (short)0, acc[i][j], false, false);
    }

    // epilogue: C/D layout -> lane = N (both halves), VGPR r = M (r or r+8)
#pragma unroll
    for (int j = 0; j < 4; ++j) {
        int e = colBase + waveN * 64 + j * 16 + (lane & 15);
        float bv = bias[e];
#pragma unroll
        for (int i = 0; i < 2; ++i) {
            int mrow = rowBase + waveM * 32 + i * 16 + (lane >> 4) * 8;
#pragma unroll
            for (int r = 0; r < 8; ++r)
                o[(size_t)(mrow + r) * EMB + e] = acc[i][j][r] + bv;
        }
    }
}

// ---------------- 5. masked per-(b,e) channel mean/std --------------------
__global__ void stats_kernel(const float* __restrict__ out,
                             const float* __restrict__ g1s, const float* __restrict__ g2s,
                             float* __restrict__ mean1, float* __restrict__ std1,
                             float* __restrict__ mean2, float* __restrict__ std2) {
    int e = blockIdx.x * blockDim.x + threadIdx.x;   // 0..767
    int b = blockIdx.y;
    const float* f1 = out + (size_t)b * PPB * EMB;
    const float* f2 = out + (size_t)OSZ + (size_t)b * PPB * EMB;
    float x0a = g1s[b*4+0], y0a = g1s[b*4+1], x2a = g1s[b*4+2], y2a = g1s[b*4+3];
    float x0b = g2s[b*4+0], y0b = g2s[b*4+1], x2b = g2s[b*4+2], y2b = g2s[b*4+3];
    float s1 = 0.f, q1 = 0.f, s2 = 0.f, q2 = 0.f;
    float n1 = 0.f, n2 = 0.f;
    for (int py = 0; py < FS; ++py) {
        float fy = (float)py;
        bool yin1 = (fy >= y0a) && (fy < y2a);
        bool yin2 = (fy >= y0b) && (fy < y2b);
        for (int px = 0; px < FS; ++px) {
            float fx = (float)px;
            if (yin1 && (fx >= x0a) && (fx < x2a)) {
                float v = f1[(py * FS + px) * EMB + e];
                s1 += v; q1 += v * v; n1 += 1.f;
            }
            if (yin2 && (fx >= x0b) && (fx < x2b)) {
                float v = f2[(py * FS + px) * EMB + e];
                s2 += v; q2 += v * v; n2 += 1.f;
            }
        }
    }
    float m1 = s1 / fmaxf(n1, 1.f);
    float m2 = s2 / fmaxf(n2, 1.f);
    float v1 = (q1 - 2.f * m1 * s1 + n1 * m1 * m1) / fmaxf(n1 - 1.f, 1.f);
    float v2 = (q2 - 2.f * m2 * s2 + n2 * m2 * m2) / fmaxf(n2 - 1.f, 1.f);
    mean1[b * EMB + e] = m1;
    std1 [b * EMB + e] = sqrtf(fmaxf(v1, 0.f));
    mean2[b * EMB + e] = m2;
    std2 [b * EMB + e] = sqrtf(fmaxf(v2, 0.f));
}

// ---------------- 6. in-place cross-image blend ---------------------------
__global__ void blend_kernel(float* __restrict__ out,
                             const float* __restrict__ g1s, const float* __restrict__ g2s,
                             const int* __restrict__ sk1, const int* __restrict__ sk2,
                             const float* __restrict__ mean1, const float* __restrict__ std1,
                             const float* __restrict__ mean2, const float* __restrict__ std2) {
    int i = blockIdx.x * blockDim.x + threadIdx.x;   // quad index: MROWS*192
    if (i >= MROWS * 192) return;
    int row = i / 192;
    int eq  = (i % 192) * 4;
    int b  = row / PPB, p = row % PPB;
    int py = p / FS,    px = p % FS;
    float fx = (float)px, fy = (float)py;
    bool in1 = (fx >= g1s[b*4+0]) && (fx < g1s[b*4+2]) && (fy >= g1s[b*4+1]) && (fy < g1s[b*4+3]);
    bool in2 = (fx >= g2s[b*4+0]) && (fx < g2s[b*4+2]) && (fy >= g2s[b*4+1]) && (fy < g2s[b*4+3]);
    bool bl1 = (!sk1[b]) && in2;   // out1 blended where mask2, unless skip1
    bool bl2 = (!sk2[b]) && in1;   // out2 blended where mask1, unless skip2

    float4* p1 = (float4*)(out + (size_t)row * EMB + eq);
    float4* p2 = (float4*)(out + (size_t)OSZ + (size_t)row * EMB + eq);
    float4 f1 = *p1;
    float4 f2 = *p2;
    if (!(bl1 || bl2)) return;

    float4 m1v = *(const float4*)(mean1 + b * EMB + eq);
    float4 s1v = *(const float4*)(std1  + b * EMB + eq);
    float4 m2v = *(const float4*)(mean2 + b * EMB + eq);
    float4 s2v = *(const float4*)(std2  + b * EMB + eq);

    if (bl1) {
        float4 o1;
        o1.x = (f2.x - m2v.x) / (s2v.x + EPSV) * (s1v.x + EPSV) + m1v.x;
        o1.y = (f2.y - m2v.y) / (s2v.y + EPSV) * (s1v.y + EPSV) + m1v.y;
        o1.z = (f2.z - m2v.z) / (s2v.z + EPSV) * (s1v.z + EPSV) + m1v.z;
        o1.w = (f2.w - m2v.w) / (s2v.w + EPSV) * (s1v.w + EPSV) + m1v.w;
        *p1 = o1;
    }
    if (bl2) {
        float4 o2;
        o2.x = (f1.x - m1v.x) / (s1v.x + EPSV) * (s2v.x + EPSV) + m2v.x;
        o2.y = (f1.y - m1v.y) / (s1v.y + EPSV) * (s2v.y + EPSV) + m2v.y;
        o2.z = (f1.z - m1v.z) / (s1v.z + EPSV) * (s2v.z + EPSV) + m2v.z;
        o2.w = (f1.w - m1v.w) / (s1v.w + EPSV) * (s2v.w + EPSV) + m2v.w;
        *p2 = o2;
    }
}

// ---------------- launcher ------------------------------------------------
extern "C" void kernel_launch(void* const* d_in, const int* in_sizes, int n_in,
                              void* d_out, int out_size, void* d_ws, size_t ws_size,
                              hipStream_t stream) {
    (void)in_sizes; (void)n_in; (void)out_size; (void)ws_size;
    const float* x    = (const float*)d_in[0];
    const float* x2   = (const float*)d_in[1];
    const float* gt1  = (const float*)d_in[2];
    const float* gt2  = (const float*)d_in[3];
    const float* W    = (const float*)d_in[4];
    const float* bias = (const float*)d_in[5];
    float* out = (float*)d_out;

    char* ws = (char*)d_ws;
    unsigned short* Wt = (unsigned short*)ws; ws += (size_t)EMB * EMB * 2;
    unsigned short* A1 = (unsigned short*)ws; ws += (size_t)MROWS * EMB * 2;
    unsigned short* A2 = (unsigned short*)ws; ws += (size_t)MROWS * EMB * 2;
    float* mean1 = (float*)ws; ws += (size_t)BATCH * EMB * 4;
    float* std1  = (float*)ws; ws += (size_t)BATCH * EMB * 4;
    float* mean2 = (float*)ws; ws += (size_t)BATCH * EMB * 4;
    float* std2  = (float*)ws; ws += (size_t)BATCH * EMB * 4;
    float* g1s   = (float*)ws; ws += (size_t)BATCH * 4 * 4;
    float* g2s   = (float*)ws; ws += (size_t)BATCH * 4 * 4;
    int*   sk1   = (int*)ws;   ws += (size_t)BATCH * 4;
    int*   sk2   = (int*)ws;   ws += (size_t)BATCH * 4;

    // 1. pack weights to bf16
    convw_kernel<<<dim3((EMB * EMB / 4 + 255) / 256), 256, 0, stream>>>(W, Wt);
    // 2. im2col both images (z via grid.y)
    im2col_kernel<<<dim3(MROWS * 192 / 256, 2), 256, 0, stream>>>(x, x2, A1, A2);
    // 3. boxes / skip flags / g1n,g2n outputs
    box_kernel<<<1, 64, 0, stream>>>(gt1, gt2, g1s, g2s, sk1, sk2,
                                     out + (size_t)2 * OSZ,
                                     out + (size_t)2 * OSZ + BATCH * 4);
    // 4. WMMA GEMM, writes o1/o2 directly in [b][p][e] layout
    gemm_kernel<<<dim3(EMB / 128, MROWS / 128, 2), 256, 0, stream>>>(A1, A2, Wt, bias, out);
    // 5. masked channel stats
    stats_kernel<<<dim3(EMB / 256, BATCH), 256, 0, stream>>>(out, g1s, g2s, mean1, std1, mean2, std2);
    // 6. in-place blend
    blend_kernel<<<dim3(MROWS * 192 / 256), 256, 0, stream>>>(out, g1s, g2s, sk1, sk2,
                                                              mean1, std1, mean2, std2);
}
